// CausalSelfAttention_78365973283219
// MI455X (gfx1250) — matrix operand order repaired and checked by reading, not compiled, not run on hardware
//
#include <hip/hip_runtime.h>

// ---------------------------------------------------------------------------
// CDNA5 (gfx1250) causal self-attention: bf16 WMMA everywhere.
//   x:[4,2048,1024] f32, w_qkv:[3072,1024] f32, w_proj:[1024,1024] f32
//   out:[4,2048,1024] f32
// Pipeline: cvt -> QKV GEMM (scatter Q,K,V^T) -> flash attention -> out proj.
// GEMMs: 32x64 tile/wave, register double-buffered fragments so L2 load
// latency overlaps the WMMA pipe instead of s_wait_loadcnt-to-zero stalls.
// ---------------------------------------------------------------------------

typedef __bf16 bf16_t;
typedef __attribute__((ext_vector_type(16))) __bf16 v16bf;
typedef __attribute__((ext_vector_type(8)))  __bf16 v8bf;
typedef __attribute__((ext_vector_type(8)))  float   v8f;
typedef __attribute__((ext_vector_type(4)))  float   v4f;

#define SEQ 2048
#define DM  1024
#define NH  16
#define HD  64
#define MTOT (4 * SEQ)       // 8192 total rows

__device__ __forceinline__ v8f zero8() {
  v8f r;
#pragma unroll
  for (int i = 0; i < 8; ++i) r[i] = 0.0f;
  return r;
}

// A-fragment (16x32 bf16, row-major source, leading dim ld).
// k order used here (not the ISA table order): lanes 0-15 row m=lane, K = 16h .. 16h+15 (contiguous), h=lane/16.
__device__ __forceinline__ v16bf load_a_frag(const bf16_t* base, int ld, int lane) {
  const int m = lane & 15, h = lane >> 4;
  const bf16_t* row = base + (size_t)m * ld + 16 * h;
  union { v16bf v; v8bf p[2]; } u;
  u.p[0] = *(const v8bf*)(row);        // K = 16h .. 16h+7
  u.p[1] = *(const v8bf*)(row + 8);    // K = 16h+8 .. 16h+15
  return u.v;
}

// B-fragment (32x16 bf16) where B[k][n] = src[n*ld + k]  (src row-major [N x K]).
// k order used here (same as load_a_frag): lane holds column n=lane%16, K = 16*(lane/16) + e (contiguous).
__device__ __forceinline__ v16bf load_bt_frag(const bf16_t* base, int ld, int lane) {
  const int n = lane & 15, h = lane >> 4;
  return *(const v16bf*)(base + (size_t)n * ld + 16 * h);
}

__device__ __forceinline__ v8f wmma_bf16(v16bf a, v16bf b, v8f c) {
  // (neg_a, A, neg_b, B, c_mod, C, reuse_a, reuse_b)
  return __builtin_amdgcn_wmma_f32_16x16x32_bf16(false, a, false, b, (short)0, c,
                                                 false, false);
}

// ---------------------------------------------------------------------------
// Kernel 1: fp32 -> bf16 conversion (vectorized x4)
// ---------------------------------------------------------------------------
__global__ void cvt_f32_to_bf16(const float* __restrict__ src,
                                bf16_t* __restrict__ dst, int n4) {
  int i = blockIdx.x * blockDim.x + threadIdx.x;
  if (i < n4) {
    v4f f = *(const v4f*)(src + 4 * (size_t)i);
    union { bf16_t b[4]; unsigned long long u; } p;
#pragma unroll
    for (int j = 0; j < 4; ++j) p.b[j] = (bf16_t)f[j];
    *(unsigned long long*)(dst + 4 * (size_t)i) = p.u;
  }
}

// ---------------------------------------------------------------------------
// Kernel 2: QKV GEMM  qkv[m][n] = sum_k x[m][k] * Wqkv[n][k]
//   M=8192, N=3072, K=1024. One wave -> 32x64 tile, double-buffered frags.
//   Epilogue scatters to Q[b,h,s,d], K[b,h,s,d], V^T[b,h,d,s] (bf16).
// ---------------------------------------------------------------------------
__global__ void __launch_bounds__(256) qkv_gemm_scatter(
    const bf16_t* __restrict__ xb, const bf16_t* __restrict__ wb,
    bf16_t* __restrict__ q, bf16_t* __restrict__ k, bf16_t* __restrict__ vt) {
  const int lane = threadIdx.x & 31;
  const int wave = blockIdx.x * 8 + (threadIdx.x >> 5);
  const int mt = wave / 48, nt = wave % 48;   // 256 x 48 tiles of 32x64
  const int m0 = mt * 32, n0 = nt * 64;

  v8f acc[2][4];
#pragma unroll
  for (int i = 0; i < 2; ++i)
#pragma unroll
    for (int t = 0; t < 4; ++t) acc[i][t] = zero8();

  const bf16_t* x0 = xb + (size_t)m0 * DM;
  const bf16_t* x1 = xb + (size_t)(m0 + 16) * DM;
  const bf16_t* w0 = wb + (size_t)n0 * DM;

  v16bf a0 = load_a_frag(x0, DM, lane);
  v16bf a1 = load_a_frag(x1, DM, lane);
  v16bf b0 = load_bt_frag(w0,                    DM, lane);
  v16bf b1 = load_bt_frag(w0 + (size_t)16 * DM,  DM, lane);
  v16bf b2 = load_bt_frag(w0 + (size_t)32 * DM,  DM, lane);
  v16bf b3 = load_bt_frag(w0 + (size_t)48 * DM,  DM, lane);

  for (int kk = 0; kk < DM; kk += 32) {
    // issue next iteration's fragment loads first (clamped in-bounds on last)
    const int kn = (kk + 32 < DM) ? kk + 32 : kk;
    v16bf na0 = load_a_frag(x0 + kn, DM, lane);
    v16bf na1 = load_a_frag(x1 + kn, DM, lane);
    v16bf nb0 = load_bt_frag(w0 + kn,                   DM, lane);
    v16bf nb1 = load_bt_frag(w0 + (size_t)16 * DM + kn, DM, lane);
    v16bf nb2 = load_bt_frag(w0 + (size_t)32 * DM + kn, DM, lane);
    v16bf nb3 = load_bt_frag(w0 + (size_t)48 * DM + kn, DM, lane);

    acc[0][0] = wmma_bf16(a0, b0, acc[0][0]);
    acc[0][1] = wmma_bf16(a0, b1, acc[0][1]);
    acc[0][2] = wmma_bf16(a0, b2, acc[0][2]);
    acc[0][3] = wmma_bf16(a0, b3, acc[0][3]);
    acc[1][0] = wmma_bf16(a1, b0, acc[1][0]);
    acc[1][1] = wmma_bf16(a1, b1, acc[1][1]);
    acc[1][2] = wmma_bf16(a1, b2, acc[1][2]);
    acc[1][3] = wmma_bf16(a1, b3, acc[1][3]);

    a0 = na0; a1 = na1; b0 = nb0; b1 = nb1; b2 = nb2; b3 = nb3;
  }

  const int h = lane >> 4, nl = lane & 15;
#pragma unroll
  for (int i = 0; i < 2; ++i) {
#pragma unroll
    for (int t = 0; t < 4; ++t) {
#pragma unroll
      for (int r = 0; r < 8; ++r) {
        const float val = acc[i][t][r];
        const int m = m0 + i * 16 + r + 8 * h;   // C layout: row = r + 8*(lane/16)
        const int n = n0 + t * 16 + nl;          //           col = lane%16
        const int b = m >> 11, s = m & (SEQ - 1);
        const int which = n >> 10, nn = n & (DM - 1);
        const int head = nn >> 6, c = nn & (HD - 1);
        const size_t hb = (size_t)(b * NH + head);
        const bf16_t bv = (bf16_t)val;
        if (which == 0)      q [(hb * SEQ + s) * HD + c] = bv;
        else if (which == 1) k [(hb * SEQ + s) * HD + c] = bv;
        else                 vt[(hb * HD + c) * SEQ + s] = bv;   // transposed V
      }
    }
  }
}

// ---------------------------------------------------------------------------
// Kernel 3: flash attention. One wave per (b, h, 16-row q-tile).
//   S = Q K^T / 8 with causal mask, online softmax, O += P V.
//   K frags double-buffered across key blocks; V frags issued right after the
//   S WMMAs so their latency hides under the softmax VALU section.
// ---------------------------------------------------------------------------
__global__ void __launch_bounds__(256) attn_flash(
    const bf16_t* __restrict__ q, const bf16_t* __restrict__ kmat,
    const bf16_t* __restrict__ vt, bf16_t* __restrict__ y) {
  __shared__ __align__(32) bf16_t pbuf[8][16 * 32];

  const int lane = threadIdx.x & 31;
  const int wid  = threadIdx.x >> 5;
  const int gw   = blockIdx.x * 8 + wid;        // [0, 8192)
  const int qt   = gw & 127;                    // q tile in [0,128)
  const int bh   = gw >> 7;                     // (b,h) in [0,64)
  const int b    = bh >> 4, head = bh & 15;
  const bf16_t* qh = q    + (size_t)bh * SEQ * HD;
  const bf16_t* kh = kmat + (size_t)bh * SEQ * HD;
  const bf16_t* vh = vt   + (size_t)bh * HD * SEQ;
  const int q0  = qt * 16;
  const int h16 = lane >> 4, nl = lane & 15;

  const v16bf qa0 = load_a_frag(qh + (size_t)q0 * HD,      HD, lane); // d 0..31
  const v16bf qa1 = load_a_frag(qh + (size_t)q0 * HD + 32, HD, lane); // d 32..63

  v8f acc0 = zero8(), acc1 = zero8(), acc2 = zero8(), acc3 = zero8();
  float rmax[8], rsum[8];
#pragma unroll
  for (int r = 0; r < 8; ++r) { rmax[r] = -3.0e38f; rsum[r] = 0.0f; }

  // K fragments for the first key block
  v16bf kf0 = load_bt_frag(kh,                        HD, lane);
  v16bf kf1 = load_bt_frag(kh + 32,                   HD, lane);
  v16bf kf2 = load_bt_frag(kh + (size_t)16 * HD,      HD, lane);
  v16bf kf3 = load_bt_frag(kh + (size_t)16 * HD + 32, HD, lane);

  bf16_t* pb = pbuf[wid];
  const int kend = q0 + 16;                      // causal bound
  for (int kb = 0; kb < kend; kb += 32) {
    // S block 16x32: two 16x16 C tiles, K(d)=64 -> 2 WMMAs each
    v8f s0 = zero8(), s1 = zero8();
    s0 = wmma_bf16(qa0, kf0, s0);
    s0 = wmma_bf16(qa1, kf1, s0);
    s1 = wmma_bf16(qa0, kf2, s1);
    s1 = wmma_bf16(qa1, kf3, s1);

    // issue V loads (this block) + K loads (next block) under the softmax
    v16bf vb0 = load_bt_frag(vh + (size_t) 0 * SEQ + kb, SEQ, lane);
    v16bf vb1 = load_bt_frag(vh + (size_t)16 * SEQ + kb, SEQ, lane);
    v16bf vb2 = load_bt_frag(vh + (size_t)32 * SEQ + kb, SEQ, lane);
    v16bf vb3 = load_bt_frag(vh + (size_t)48 * SEQ + kb, SEQ, lane);
    const int kbn = (kb + 32 < kend) ? kb + 32 : kb;   // clamp: stays in bounds
    v16bf nk0 = load_bt_frag(kh + (size_t)kbn * HD,             HD, lane);
    v16bf nk1 = load_bt_frag(kh + (size_t)kbn * HD + 32,        HD, lane);
    v16bf nk2 = load_bt_frag(kh + (size_t)(kbn + 16) * HD,      HD, lane);
    v16bf nk3 = load_bt_frag(kh + (size_t)(kbn + 16) * HD + 32, HD, lane);

    // scale + causal mask + per-lane row max
    const bool need_mask = (kb + 32 > q0);
    float p0[8], p1[8], mx[8], corr[8];
#pragma unroll
    for (int r = 0; r < 8; ++r) {
      float v0 = s0[r] * 0.125f;                 // 1/sqrt(64)
      float v1 = s1[r] * 0.125f;
      if (need_mask) {
        const int m = q0 + r + 8 * h16;
        if (kb + nl      > m) v0 = -3.0e38f;
        if (kb + 16 + nl > m) v1 = -3.0e38f;
      }
      p0[r] = v0; p1[r] = v1;
      mx[r] = fmaxf(v0, v1);
    }
    // row max across the 16 lanes holding one row (xor 1..8 stays in half)
#pragma unroll
    for (int off = 1; off < 16; off <<= 1)
#pragma unroll
      for (int r = 0; r < 8; ++r)
        mx[r] = fmaxf(mx[r], __shfl_xor(mx[r], off, 32));

#pragma unroll
    for (int r = 0; r < 8; ++r) {
      const float mnew = fmaxf(rmax[r], mx[r]);
      corr[r] = __expf(rmax[r] - mnew);
      rmax[r] = mnew;
      p0[r] = __expf(p0[r] - mnew);
      p1[r] = __expf(p1[r] - mnew);
    }
#pragma unroll
    for (int r = 0; r < 8; ++r) {
      acc0[r] *= corr[r]; acc1[r] *= corr[r];
      acc2[r] *= corr[r]; acc3[r] *= corr[r];
    }
    float ps[8];
#pragma unroll
    for (int r = 0; r < 8; ++r) ps[r] = p0[r] + p1[r];
#pragma unroll
    for (int off = 1; off < 16; off <<= 1)
#pragma unroll
      for (int r = 0; r < 8; ++r)
        ps[r] += __shfl_xor(ps[r], off, 32);
#pragma unroll
    for (int r = 0; r < 8; ++r) rsum[r] = rsum[r] * corr[r] + ps[r];

    // stage P (C layout -> row-major 16x32 bf16) in LDS, reload as A fragment
#pragma unroll
    for (int r = 0; r < 8; ++r) {
      pb[(r + 8 * h16) * 32 + nl]      = (bf16_t)p0[r];
      pb[(r + 8 * h16) * 32 + 16 + nl] = (bf16_t)p1[r];
    }
    asm volatile("s_wait_dscnt 0" ::: "memory");   // DS in-order per wave; fence compiler
    const v16bf pa = load_a_frag(pb, 32, lane);

    // O += P V : B-frags from V^T rows (contiguous along keys)
    acc0 = wmma_bf16(pa, vb0, acc0);
    acc1 = wmma_bf16(pa, vb1, acc1);
    acc2 = wmma_bf16(pa, vb2, acc2);
    acc3 = wmma_bf16(pa, vb3, acc3);

    kf0 = nk0; kf1 = nk1; kf2 = nk2; kf3 = nk3;
  }

  // epilogue: normalize and write Y[b, s, h*64 + c] (bf16, row-major for proj)
  v8f accs[4] = {acc0, acc1, acc2, acc3};
#pragma unroll
  for (int r = 0; r < 8; ++r) {
    const float inv = 1.0f / rsum[r];
    const int m = q0 + r + 8 * h16;
    bf16_t* yrow = y + ((size_t)b * SEQ + m) * DM + head * HD;
#pragma unroll
    for (int t = 0; t < 4; ++t)
      yrow[t * 16 + nl] = (bf16_t)(accs[t][r] * inv);
  }
}

// ---------------------------------------------------------------------------
// Kernel 4: output projection  out[m][n] = sum_k y[m][k] * Wproj[n][k]
//   M=8192, N=1024, K=1024; 32x64 tile/wave, double-buffered; fp32 to d_out.
// ---------------------------------------------------------------------------
__global__ void __launch_bounds__(256) proj_gemm(
    const bf16_t* __restrict__ yb, const bf16_t* __restrict__ wp,
    float* __restrict__ out) {
  const int lane = threadIdx.x & 31;
  const int wave = blockIdx.x * 8 + (threadIdx.x >> 5);
  const int mt = wave >> 4, nt = wave & 15;    // 256 x 16 tiles of 32x64
  const int m0 = mt * 32, n0 = nt * 64;

  v8f acc[2][4];
#pragma unroll
  for (int i = 0; i < 2; ++i)
#pragma unroll
    for (int t = 0; t < 4; ++t) acc[i][t] = zero8();

  const bf16_t* y0 = yb + (size_t)m0 * DM;
  const bf16_t* y1 = yb + (size_t)(m0 + 16) * DM;
  const bf16_t* w0 = wp + (size_t)n0 * DM;

  v16bf a0 = load_a_frag(y0, DM, lane);
  v16bf a1 = load_a_frag(y1, DM, lane);
  v16bf b0 = load_bt_frag(w0,                   DM, lane);
  v16bf b1 = load_bt_frag(w0 + (size_t)16 * DM, DM, lane);
  v16bf b2 = load_bt_frag(w0 + (size_t)32 * DM, DM, lane);
  v16bf b3 = load_bt_frag(w0 + (size_t)48 * DM, DM, lane);

  for (int kk = 0; kk < DM; kk += 32) {
    const int kn = (kk + 32 < DM) ? kk + 32 : kk;
    v16bf na0 = load_a_frag(y0 + kn, DM, lane);
    v16bf na1 = load_a_frag(y1 + kn, DM, lane);
    v16bf nb0 = load_bt_frag(w0 + kn,                   DM, lane);
    v16bf nb1 = load_bt_frag(w0 + (size_t)16 * DM + kn, DM, lane);
    v16bf nb2 = load_bt_frag(w0 + (size_t)32 * DM + kn, DM, lane);
    v16bf nb3 = load_bt_frag(w0 + (size_t)48 * DM + kn, DM, lane);

    acc[0][0] = wmma_bf16(a0, b0, acc[0][0]);
    acc[0][1] = wmma_bf16(a0, b1, acc[0][1]);
    acc[0][2] = wmma_bf16(a0, b2, acc[0][2]);
    acc[0][3] = wmma_bf16(a0, b3, acc[0][3]);
    acc[1][0] = wmma_bf16(a1, b0, acc[1][0]);
    acc[1][1] = wmma_bf16(a1, b1, acc[1][1]);
    acc[1][2] = wmma_bf16(a1, b2, acc[1][2]);
    acc[1][3] = wmma_bf16(a1, b3, acc[1][3]);

    a0 = na0; a1 = na1; b0 = nb0; b1 = nb1; b2 = nb2; b3 = nb3;
  }

  const int h = lane >> 4, nl = lane & 15;
#pragma unroll
  for (int i = 0; i < 2; ++i) {
#pragma unroll
    for (int r = 0; r < 8; ++r) {
      const int m = m0 + i * 16 + r + 8 * h;
      float* orow = out + (size_t)m * DM + n0;
#pragma unroll
      for (int t = 0; t < 4; ++t)
        orow[t * 16 + nl] = acc[i][t][r];       // lanes 0..15 -> consecutive n
    }
  }
}

// ---------------------------------------------------------------------------
extern "C" void kernel_launch(void* const* d_in, const int* in_sizes, int n_in,
                              void* d_out, int out_size, void* d_ws, size_t ws_size,
                              hipStream_t stream) {
  (void)in_sizes; (void)n_in; (void)out_size; (void)ws_size;
  const float* x  = (const float*)d_in[0];   // [4,2048,1024]
  const float* wq = (const float*)d_in[1];   // [3072,1024]
  const float* wp = (const float*)d_in[2];   // [1024,1024]
  float* out = (float*)d_out;

  // Workspace carve-out (256B aligned). Y reuses the x_bf16 buffer (dead after QKV).
  char* ws = (char*)d_ws;
  size_t off = 0;
  auto carve = [&](size_t bytes) {
    char* p = ws + off;
    off += (bytes + 255) & ~(size_t)255;
    return p;
  };
  bf16_t* xb  = (bf16_t*)carve((size_t)MTOT * DM * sizeof(bf16_t));     // 16 MB (also Y)
  bf16_t* wqb = (bf16_t*)carve((size_t)3 * DM * DM * sizeof(bf16_t));   //  6 MB
  bf16_t* wpb = (bf16_t*)carve((size_t)DM * DM * sizeof(bf16_t));       //  2 MB
  bf16_t* qb  = (bf16_t*)carve((size_t)MTOT * DM * sizeof(bf16_t));     // 16 MB
  bf16_t* kb  = (bf16_t*)carve((size_t)MTOT * DM * sizeof(bf16_t));     // 16 MB
  bf16_t* vtb = (bf16_t*)carve((size_t)MTOT * DM * sizeof(bf16_t));     // 16 MB
  bf16_t* yb  = xb;                                                     // alias

  int n4 = (int)((size_t)MTOT * DM / 4);
  cvt_f32_to_bf16<<<(n4 + 255) / 256, 256, 0, stream>>>(x, xb, n4);
  n4 = 3 * DM * DM / 4;
  cvt_f32_to_bf16<<<(n4 + 255) / 256, 256, 0, stream>>>(wq, wqb, n4);
  n4 = DM * DM / 4;
  cvt_f32_to_bf16<<<(n4 + 255) / 256, 256, 0, stream>>>(wp, wpb, n4);

  // 256*48 tiles of 32x64, 8 waves/block
  qkv_gemm_scatter<<<1536, 256, 0, stream>>>(xb, wqb, qb, kb, vtb);
  // 64 (b,h) * 128 q-tiles = 8192 waves
  attn_flash<<<1024, 256, 0, stream>>>(qb, kb, vtb, yb);
  // 256*16 tiles of 32x64
  proj_gemm<<<512, 256, 0, stream>>>(yb, wpb, out);
}
